// MSLA_3015067042154
// MI455X (gfx1250) — compile-verified, matrix operand order checked by reading, not run on hardware
//
#include <hip/hip_runtime.h>
#include <hip/hip_bf16.h>

// ---------------------------------------------------------------------------
// MSLA (multi-scale linear attention) for MI455X / gfx1250, wave32 + WMMA.
// All GEMM operands are staged as native f16 in workspace; global->LDS tile
// staging uses GLOBAL_LOAD_ASYNC_TO_LDS_B128 (ASYNCcnt) with double-buffered
// LDS so chunk i+1's DMA overlaps chunk i's WMMAs. Each wave runs a 2x2 grid
// of v_wmma_f32_16x16x32_f16.
// ---------------------------------------------------------------------------

typedef __attribute__((ext_vector_type(16))) _Float16 v16h;
typedef __attribute__((ext_vector_type(8)))  _Float16 v8h;
typedef __attribute__((ext_vector_type(8)))  float    v8f;
typedef __attribute__((ext_vector_type(4)))  int      v4i;

#define NB    16
#define CC    64
#define NPIX  6400            // 80*80
#define MROWS (NB * NPIX)     // 102400 = 800 * 128

#if defined(__has_builtin)
#if __has_builtin(__builtin_amdgcn_global_load_async_to_lds_b128) && \
    __has_builtin(__builtin_amdgcn_s_wait_asynccnt)
#define USE_ASYNC_LDS 1
#endif
#endif
#ifndef USE_ASYNC_LDS
#define USE_ASYNC_LDS 0
#endif

// 16-byte global -> LDS tile copy (8 f16) via async DMA (ASYNCcnt-tracked).
__device__ __forceinline__ void copy8_g2l(const _Float16* g, _Float16* l)
{
#if USE_ASYNC_LDS
    __builtin_amdgcn_global_load_async_to_lds_b128(
        (__attribute__((address_space(1))) v4i*)g,
        (__attribute__((address_space(3))) v4i*)l, 0, 0);
#else
    *(v8h*)l = *(const v8h*)g;
#endif
}

// Wait until this wave's outstanding async copies <= N (in-order completion).
template <int N>
__device__ __forceinline__ void async_wait_le()
{
#if USE_ASYNC_LDS
    __builtin_amdgcn_s_wait_asynccnt(N);
#endif
}

// ---------------------------------------------------------------------------
// 0) one-shot f32 -> f16 weight conversion
// ---------------------------------------------------------------------------
__global__ void f32_to_f16_kernel(const float* __restrict__ src,
                                  _Float16* __restrict__ dst, int n)
{
    int i = blockIdx.x * 256 + threadIdx.x;
    if (i < n) dst[i] = (_Float16)src[i];
}

// ---------------------------------------------------------------------------
// 1) depthwise conv + bias + residual + relu; writes transposed f16 y^T[b,n,c]
// ---------------------------------------------------------------------------
template <int KSZ>
__global__ void dwconv_kernel(const float* __restrict__ x,   // branch base, batch stride 256*6400
                              const float* __restrict__ w,   // [64, KSZ, KSZ]
                              const float* __restrict__ bias,
                              _Float16* __restrict__ yt)     // [b, 6400, 64] f16
{
    int id = blockIdx.x * blockDim.x + threadIdx.x;          // NB*CC*NPIX threads
    int hw = id % NPIX;
    int c  = (id / NPIX) % CC;
    int b  = id / (NPIX * CC);
    int hh = hw / 80, ww = hw % 80;
    const float* xb = x + (size_t)b * (256 * NPIX) + (size_t)c * NPIX;
    const float* wc = w + c * KSZ * KSZ;
    const int P = KSZ / 2;
    float acc = bias[c] + xb[hw];
#pragma unroll
    for (int i = 0; i < KSZ; ++i) {
        int yy = hh + i - P;
        if (yy < 0 || yy >= 80) continue;
#pragma unroll
        for (int j = 0; j < KSZ; ++j) {
            int xx = ww + j - P;
            if (xx < 0 || xx >= 80) continue;
            acc += xb[yy * 80 + xx] * wc[i * KSZ + j];
        }
    }
    yt[(size_t)b * NPIX * CC + (size_t)hw * CC + c] = (_Float16)fmaxf(acc, 0.0f);
}

// ---------------------------------------------------------------------------
// WMMA fragment load from an LDS tile row-major [rows][32] f16.
// CDNA5 16-bit A/B layout: lane l holds row (l&15); lanes 0-15 carry
// K = {0..7, 16..23}, lanes 16-31 carry K = {8..15, 24..31}; each half is 8
// contiguous f16 -> two ds_load_b128 per fragment.
// ---------------------------------------------------------------------------
__device__ __forceinline__ v16h load_frag(const _Float16* base, int lane)
{
    int r  = lane & 15;
    int hi = lane >> 4;
    const _Float16* p = base + r * 32 + hi * 8;
    v8h lo = *(const v8h*)(p);       // K = hi*8 .. +7
    v8h up = *(const v8h*)(p + 16);  // K = 16+hi*8 .. +7
    v16h f;
#pragma unroll
    for (int i = 0; i < 8; ++i) { f[i] = lo[i]; f[i + 8] = up[i]; }
    return f;
}

__device__ __forceinline__ v8f wmma_f16(v16h a, v16h b, v8f c)
{
    return __builtin_amdgcn_wmma_f32_16x16x32_f16(false, a, false, b,
                                                  (short)0, c, false, false);
}

// ---------------------------------------------------------------------------
// Generic WMMA GEMM: out = A[M,K](f16) x W[N,K]^T(f16) (+bias)(*scale)
//   grid = (M/128, N/64), block = 256 (8 waves).
//   Wave w owns a 32x32 patch: mSub=(w&3)*32, nSub=(w>>2)*32 -> 4 WMMAs/chunk.
//   Double-buffered LDS: chunk i+1 DMA overlaps chunk i WMMAs.
// mode 0: outF[m*192 + col]                        (qkv, no bias, f32 out)
// mode 1: outH[m*256 + brOff + col] = (acc+b)*s    (proj into concat, f16 out)
// mode 2: outF[(b*256+col)*6400 + n] = acc + b     (fuse, transposed f32 out)
// ---------------------------------------------------------------------------
__global__ void wmma_gemm(const _Float16* __restrict__ A,
                          const _Float16* __restrict__ W,
                          const float* __restrict__ bias,
                          const float* __restrict__ scalep,
                          float* __restrict__ outF,
                          _Float16* __restrict__ outH,
                          int Ktot, int mode, int brOff)
{
    __shared__ __attribute__((aligned(16))) _Float16 As[2][128 * 32];
    __shared__ __attribute__((aligned(16))) _Float16 Bs[2][64 * 32];

    const int tid  = threadIdx.x;
    const int lane = tid & 31;
    const int wave = tid >> 5;
    const int mSub = (wave & 3) * 32;            // 0,32,64,96
    const int nSub = (wave >> 2) * 32;           // 0 or 32
    const size_t mTile = (size_t)blockIdx.x * 128;
    const int    nTile = blockIdx.y * 64;

    // cooperative copy mapping (b128 granularity)
    const int aRow0 = tid >> 2;                  // A slots tid, tid+256
    const int aCol  = (tid & 3) * 8;
    const int bRow  = tid >> 2;                  // B slot tid
    const int bCol  = (tid & 3) * 8;

    v8f acc00 = {}, acc01 = {}, acc10 = {}, acc11 = {};

    // prologue: stage chunk 0 into buffer 0 (3 async copies per thread)
    {
        copy8_g2l(A + (mTile + aRow0) * (size_t)Ktot + aCol,       &As[0][aRow0 * 32 + aCol]);
        copy8_g2l(A + (mTile + aRow0 + 64) * (size_t)Ktot + aCol,  &As[0][(aRow0 + 64) * 32 + aCol]);
        copy8_g2l(W + (size_t)(nTile + bRow) * Ktot + bCol,        &Bs[0][bRow * 32 + bCol]);
    }

    const int nChunks = Ktot >> 5;               // 2 (K=64) or 8 (K=256)
    for (int c = 0; c < nChunks; ++c) {
        const int cur = c & 1;
        if (c + 1 < nChunks) {
            const int kn = (c + 1) * 32;
            const int nxt = cur ^ 1;
            const _Float16* ga = A + (mTile + aRow0) * (size_t)Ktot + kn + aCol;
            if (c + 2 < nChunks) __builtin_prefetch(ga + 32, 0, 1);
            copy8_g2l(ga,                                               &As[nxt][aRow0 * 32 + aCol]);
            copy8_g2l(A + (mTile + aRow0 + 64) * (size_t)Ktot + kn + aCol,
                                                                        &As[nxt][(aRow0 + 64) * 32 + aCol]);
            copy8_g2l(W + (size_t)(nTile + bRow) * Ktot + kn + bCol,    &Bs[nxt][bRow * 32 + bCol]);
            async_wait_le<3>();                  // chunk c's 3 copies done; c+1 in flight
        } else {
            async_wait_le<0>();                  // final chunk: drain
        }
        __syncthreads();

        v16h a0 = load_frag(&As[cur][(mSub     ) * 32], lane);
        v16h a1 = load_frag(&As[cur][(mSub + 16) * 32], lane);
        v16h b0 = load_frag(&Bs[cur][(nSub     ) * 32], lane);
        v16h b1 = load_frag(&Bs[cur][(nSub + 16) * 32], lane);
        acc00 = wmma_f16(a0, b0, acc00);
        acc01 = wmma_f16(a0, b1, acc01);
        acc10 = wmma_f16(a1, b0, acc10);
        acc11 = wmma_f16(a1, b1, acc11);
        __syncthreads();                         // LDS buffer reuse fence
    }

    // D layout: VGPR r -> M = r (lanes 0-15) / r+8 (lanes 16-31), N = lane&15
    const int   mOff = (lane >> 4) * 8;
    const float sv   = scalep ? scalep[0] : 1.0f;

#pragma unroll
    for (int mi = 0; mi < 2; ++mi) {
#pragma unroll
        for (int ni = 0; ni < 2; ++ni) {
            const v8f acc = mi ? (ni ? acc11 : acc10) : (ni ? acc01 : acc00);
            const int colG = nTile + nSub + ni * 16 + (lane & 15);
            const float bv = bias ? bias[colG] : 0.0f;
#pragma unroll
            for (int r = 0; r < 8; ++r) {
                const size_t m = mTile + mSub + mi * 16 + mOff + r;
                float v = acc[r] + bv;
                if (mode == 0) {
                    outF[m * 192 + colG] = v;
                } else if (mode == 1) {
                    outH[m * 256 + brOff + colG] = (_Float16)(v * sv);
                } else {
                    const size_t b  = m / NPIX;
                    const size_t nn = m - b * NPIX;
                    outF[(b * 256 + colG) * NPIX + nn] = v;
                }
            }
        }
    }
}

// ---------------------------------------------------------------------------
// 3) softmax over head-dim (8 contiguous floats) on the K slice
// ---------------------------------------------------------------------------
__global__ void ksoftmax_kernel(float* __restrict__ qkv, int total)
{
    int id = blockIdx.x * blockDim.x + threadIdx.x;
    if (id >= total) return;                     // total = MROWS*8
    float* p = qkv + (size_t)(id >> 3) * 192 + 64 + (id & 7) * 8;
    float mx = p[0];
#pragma unroll
    for (int i = 1; i < 8; ++i) mx = fmaxf(mx, p[i]);
    float e[8], s = 0.0f;
#pragma unroll
    for (int i = 0; i < 8; ++i) { e[i] = __expf(p[i] - mx); s += e[i]; }
    float inv = 1.0f / s;
#pragma unroll
    for (int i = 0; i < 8; ++i) p[i] = e[i] * inv;
}

// ---------------------------------------------------------------------------
// 4) softmax over tokens (6400, stride 192) on the Q slice; block per (b,h,d)
// ---------------------------------------------------------------------------
__global__ void qsoftmax_kernel(float* __restrict__ qkv)
{
    __shared__ float red[256];
    const int b  = blockIdx.x >> 6;              // grid = 16*64
    const int hd = blockIdx.x & 63;              // h*8 + d
    float* base  = qkv + (size_t)b * NPIX * 192 + hd;
    const int t  = threadIdx.x;

    float mx = -3.4e38f;
    for (int nn = t; nn < NPIX; nn += 256) mx = fmaxf(mx, base[(size_t)nn * 192]);
    red[t] = mx; __syncthreads();
    for (int s = 128; s > 0; s >>= 1) { if (t < s) red[t] = fmaxf(red[t], red[t + s]); __syncthreads(); }
    mx = red[0]; __syncthreads();

    float sum = 0.0f;
    for (int nn = t; nn < NPIX; nn += 256) sum += __expf(base[(size_t)nn * 192] - mx);
    red[t] = sum; __syncthreads();
    for (int s = 128; s > 0; s >>= 1) { if (t < s) red[t] += red[t + s]; __syncthreads(); }
    const float inv = 1.0f / red[0];

    for (int nn = t; nn < NPIX; nn += 256) {
        size_t i = (size_t)nn * 192;
        base[i] = __expf(base[i] - mx) * inv;
    }
}

// ---------------------------------------------------------------------------
// 5) ctx[b,h,d,e] = sum_n k[b,h,n,d] * v[b,h,n,e]   (block per (b,h), 64 thr;
//    qkv branch buffer is L2-resident on MI455X: 78 MB < 192 MB)
// ---------------------------------------------------------------------------
__global__ void ctx_kernel(const float* __restrict__ qkv, float* __restrict__ ctx)
{
    const int b = blockIdx.x >> 3, h = blockIdx.x & 7;
    const int d = threadIdx.x >> 3, e = threadIdx.x & 7;
    const float* base = qkv + (size_t)b * NPIX * 192;
    float acc = 0.0f;
    for (int nn = 0; nn < NPIX; ++nn) {
        const float* tok = base + (size_t)nn * 192;
        acc += tok[64 + h * 8 + d] * tok[128 + h * 8 + e];
    }
    ctx[((size_t)blockIdx.x * 8 + d) * 8 + e] = acc;   // [b,h,d,e]
}

// ---------------------------------------------------------------------------
// 6) out[b,h,n,e] = sum_d q[b,h,n,d] * ctx[b,h,d,e]; f16 flat write replicates
//    torch's raw reshape [b,heads,n,dh] -> [b,n,c].
// ---------------------------------------------------------------------------
__global__ void qctx_kernel(const float* __restrict__ qkv,
                            const float* __restrict__ ctx,
                            _Float16* __restrict__ attn)
{
    int id = blockIdx.x * blockDim.x + threadIdx.x;      // NB*8*NPIX
    int nn = id % NPIX;
    int h  = (id / NPIX) & 7;
    int b  = id / (NPIX * 8);
    const float* q  = qkv + ((size_t)b * NPIX + nn) * 192 + h * 8;
    const float* cx = ctx + ((size_t)b * 8 + h) * 64;    // [d,e]
    _Float16* dst = attn + (size_t)b * (8 * NPIX * 8) + (size_t)h * (NPIX * 8) + (size_t)nn * 8;
#pragma unroll
    for (int e = 0; e < 8; ++e) {
        float a = 0.0f;
#pragma unroll
        for (int d = 0; d < 8; ++d) a += q[d] * cx[d * 8 + e];
        dst[e] = (_Float16)a;
    }
}

// ---------------------------------------------------------------------------
extern "C" void kernel_launch(void* const* d_in, const int* in_sizes, int n_in,
                              void* d_out, int out_size, void* d_ws, size_t ws_size,
                              hipStream_t stream)
{
    (void)in_sizes; (void)n_in; (void)out_size; (void)ws_size;
    const float* x     = (const float*)d_in[0];
    const float* wqkv  = (const float*)d_in[9];
    const float* wproj = (const float*)d_in[10];
    const float* bproj = (const float*)d_in[11];
    const float* wf    = (const float*)d_in[12];
    const float* bf    = (const float*)d_in[13];
    const float* scale = (const float*)d_in[14];

    // workspace layout: cat16 | qkv(f32) | yt16(=attn16, aliased) | ctx | w16s
    char* p = (char*)d_ws;
    _Float16* cat16 = (_Float16*)p;  p += (size_t)MROWS * 256 * sizeof(_Float16);
    float*    qkvb  = (float*)p;     p += (size_t)MROWS * 192 * sizeof(float);
    _Float16* yt16  = (_Float16*)p;  p += (size_t)MROWS * CC * sizeof(_Float16);
    float*    ctxb  = (float*)p;     p += (size_t)NB * 8 * 64 * sizeof(float);
    _Float16* wq16  = (_Float16*)p;  p += (size_t)192 * 64 * sizeof(_Float16);
    _Float16* wp16  = (_Float16*)p;  p += (size_t)64 * 64 * sizeof(_Float16);
    _Float16* wf16  = (_Float16*)p;

    // one-shot weight conversion to f16
    f32_to_f16_kernel<<<(192 * 64 + 255) / 256, 256, 0, stream>>>(wqkv, wq16, 192 * 64);
    f32_to_f16_kernel<<<(64 * 64 + 255) / 256, 256, 0, stream>>>(wproj, wp16, 64 * 64);
    f32_to_f16_kernel<<<(256 * 256 + 255) / 256, 256, 0, stream>>>(wf, wf16, 256 * 256);

    const int convBlocks = (NB * CC * NPIX) / 256;               // 25600
    const dim3 gQkv (MROWS / 128, 3);
    const dim3 gProj(MROWS / 128, 1);
    const dim3 gFuse(MROWS / 128, 4);

    for (int br = 0; br < 4; ++br) {
        const float* wk = (const float*)d_in[1 + br * 2];
        const float* bk = (const float*)d_in[2 + br * 2];
        const float* xb = x + (size_t)br * CC * NPIX;
        switch (br) {
            case 0: dwconv_kernel<3><<<convBlocks, 256, 0, stream>>>(xb, wk, bk, yt16); break;
            case 1: dwconv_kernel<5><<<convBlocks, 256, 0, stream>>>(xb, wk, bk, yt16); break;
            case 2: dwconv_kernel<7><<<convBlocks, 256, 0, stream>>>(xb, wk, bk, yt16); break;
            case 3: dwconv_kernel<9><<<convBlocks, 256, 0, stream>>>(xb, wk, bk, yt16); break;
        }
        wmma_gemm<<<gQkv, 256, 0, stream>>>(yt16, wq16, nullptr, nullptr,
                                            qkvb, nullptr, 64, 0, 0);
        ksoftmax_kernel<<<(MROWS * 8) / 256, 256, 0, stream>>>(qkvb, MROWS * 8);
        qsoftmax_kernel<<<NB * 64, 256, 0, stream>>>(qkvb);
        ctx_kernel<<<NB * 8, 64, 0, stream>>>(qkvb, ctxb);
        qctx_kernel<<<(NB * 8 * NPIX) / 256, 256, 0, stream>>>(qkvb, ctxb, yt16); // yt16 -> attn16
        wmma_gemm<<<gProj, 256, 0, stream>>>(yt16, wp16, bproj, scale + br,
                                             nullptr, cat16, 64, 1, br * 64);
    }
    wmma_gemm<<<gFuse, 256, 0, stream>>>(cat16, wf16, bf, nullptr,
                                         (float*)d_out, nullptr, 256, 2, 0);
}
